// GRU_43112881717881
// MI455X (gfx1250) — matrix operand order repaired and checked by reading, not compiled, not run on hardware
//
#include <hip/hip_runtime.h>
#include <hip/hip_bf16.h>

// Problem dims (fixed by the reference)
#define SEQ_LEN  512
#define BATCH    128
#define INPUT_SZ 64
#define HIDDEN   1024
#define LAYERS   2

typedef __attribute__((ext_vector_type(16))) __bf16 v16bf;
typedef __attribute__((ext_vector_type(8)))  __bf16 v8bf;
typedef __attribute__((ext_vector_type(8)))  float  v8f;

// ---------------------------------------------------------------------------
// WMMA fragment loaders for D = A(16x32) * B(32x16) + C, bf16 in / f32 out.
//
// A (16x32, MxK, 16-bit): lanes 0-15 hold row M=lane with K = k..k+7 (VGPR0-3)
// and K = k+16..k+23 (VGPR4-7); lanes 16-31 hold row M=lane-16 with K = k+8..15
// and k+24..31.  => two contiguous 16B loads per lane from row-major A.
//
// B (32x16, KxN, 16-bit): lanes 0-15 hold column N=lane with K = k..k+7 and
// k+16..k+23; lanes 16-31 hold K = k+8..15 and k+24..31 (same as A).  For
// y = x*W^T with W row-major [N,K], B column n == W row n => two 16B loads per lane.
// ---------------------------------------------------------------------------
static __device__ __forceinline__ v16bf load_a_frag(const __bf16* __restrict__ A,
                                                    int lda, int row_base, int k, int lane) {
    const int row  = row_base + (lane & 15);
    const int kofs = (lane & 16) ? 8 : 0;
    const __bf16* p = A + (size_t)row * lda + (k + kofs);
    v8bf lo = *(const v8bf*)(p);
    v8bf hi = *(const v8bf*)(p + 16);
    return __builtin_shufflevector(lo, hi, 0,1,2,3,4,5,6,7,8,9,10,11,12,13,14,15);
}

static __device__ __forceinline__ v16bf load_b_frag(const __bf16* __restrict__ W,
                                                    int ldw, int col_base, int k, int lane) {
    const int col  = col_base + (lane & 15);
    const int kofs = (lane & 16) ? 8 : 0;
    const __bf16* p = W + (size_t)col * ldw + (k + kofs);
    v8bf lo = *(const v8bf*)(p);
    v8bf hi = *(const v8bf*)(p + 16);
    return __builtin_shufflevector(lo, hi, 0,1,2,3,4,5,6,7,8,9,10,11,12,13,14,15);
}

static __device__ __forceinline__ v8f wmma_bf16(v16bf a, v16bf b, v8f c) {
    return __builtin_amdgcn_wmma_f32_16x16x32_bf16(false, a, false, b, (short)0, c, false, false);
}

static __device__ __forceinline__ float sigmoidf_fast(float x) {
    return 1.0f / (1.0f + __expf(-x));
}

// ---------------------------------------------------------------------------
// Generic C = A[M,K] * W[N,K]^T + bias with 2x2 register blocking:
// one wave owns a 32x32 output block (4 WMMA tiles), so each K-step does
// 4 WMMAs per 8 fragment loads (vs 1 per 4 unblocked) -> 2x arithmetic
// intensity on the VMEM path. 4 waves / 128-thread block.
// Requires M % 32 == 0, N % 32 == 0 (true for all uses here).
// ---------------------------------------------------------------------------
__global__ __launch_bounds__(128) void gemm_bias_wmma(
    const __bf16* __restrict__ A, const __bf16* __restrict__ W,
    const float* __restrict__ bias,
    __bf16* __restrict__ out_bf, float* __restrict__ out_f32,
    int M, int N, int K)
{
    const int lane   = threadIdx.x & 31;
    const int wave   = threadIdx.x >> 5;
    const int tile   = blockIdx.x * 4 + wave;
    const int ntiles = N >> 5;                 // 32-wide super-tiles
    const int tm = tile / ntiles;
    const int tn = tile % ntiles;
    const int row_base = tm << 5;
    const int col_base = tn << 5;

    const int n0 = col_base + (lane & 15);
    const int n1 = n0 + 16;

    v8f acc00, acc01, acc10, acc11;
    const float b0 = bias ? bias[n0] : 0.0f;
    const float b1 = bias ? bias[n1] : 0.0f;
    #pragma unroll
    for (int i = 0; i < 8; ++i) { acc00[i] = b0; acc01[i] = b1; acc10[i] = b0; acc11[i] = b1; }

    for (int k = 0; k < K; k += 32) {
        v16bf a0 = load_a_frag(A, K, row_base,      k, lane);
        v16bf a1 = load_a_frag(A, K, row_base + 16, k, lane);
        v16bf w0 = load_b_frag(W, K, col_base,      k, lane);
        v16bf w1 = load_b_frag(W, K, col_base + 16, k, lane);
        acc00 = wmma_bf16(a0, w0, acc00);
        acc01 = wmma_bf16(a0, w1, acc01);
        acc10 = wmma_bf16(a1, w0, acc10);
        acc11 = wmma_bf16(a1, w1, acc11);
    }

    // C/D layout: lane L holds column n=L%16, rows v + 8*(L>=16) for VGPR v.
    const int mrow0 = row_base + ((lane & 16) ? 8 : 0);
    #pragma unroll
    for (int v = 0; v < 8; ++v) {
        const size_t i00 = (size_t)(mrow0 + v) * N + n0;
        const size_t i10 = (size_t)(mrow0 + 16 + v) * N + n0;
        if (out_f32) {
            out_f32[i00]      = acc00[v];
            out_f32[i00 + 16] = acc01[v];
            out_f32[i10]      = acc10[v];
            out_f32[i10 + 16] = acc11[v];
        }
        if (out_bf) {
            out_bf[i00]      = (__bf16)acc00[v];
            out_bf[i00 + 16] = (__bf16)acc01[v];
            out_bf[i10]      = (__bf16)acc10[v];
            out_bf[i10 + 16] = (__bf16)acc11[v];
        }
    }
}

// ---------------------------------------------------------------------------
// One GRU time-step, K-split across the 4 waves of a block.
// Each block owns one 16x16 (batch x hidden) output tile; wave w accumulates
// K in [w*256, (w+1)*256) for all 3 gates (A fragment reused 3x per K-step).
// Waves 1-3 spill partial accumulators to LDS; after the barrier wave 0
// reduces and runs the fused sigmoid/tanh/blend gate math in-register.
// 512 blocks * 4 waves = 2048 waves in flight per step (vs 512 unsplit),
// and each wave's WMMA dependency chain is 4x shorter -- this matters because
// the 1024 sequential steps are the latency-critical path of the whole net.
// ---------------------------------------------------------------------------
#define KSPLIT (HIDDEN / 4)   // 256

__global__ __launch_bounds__(128) void gru_step_wmma(
    const __bf16* __restrict__ h_bf,    // [B,H] bf16 state (GEMM operand)
    const float*  __restrict__ h_f32,   // [B,H] fp32 state (blend operand)
    const __bf16* __restrict__ xp_t,    // [B,3H] precomputed x-projection @ t
    const __bf16* __restrict__ w_hh,    // [3H,H] bf16, gate order r,z,n
    const float*  __restrict__ b_hh,    // [3H]
    float*  __restrict__ h_f32_out,     // [B,H]
    __bf16* __restrict__ h_bf_out,      // [B,H]
    __bf16* __restrict__ y_bf)          // [B,H] layer output slice at t
{
    const int lane = threadIdx.x & 31;
    const int wave = threadIdx.x >> 5;
    const int tile = blockIdx.x;                   // 8*64 = 512 tiles
    const int ntiles = HIDDEN >> 4;                // 64
    const int tm = tile / ntiles;
    const int tn = tile % ntiles;
    const int row_base = tm << 4;
    const int col_base = tn << 4;
    const int j = col_base + (lane & 15);

    v8f accr = {}, accz = {}, accn = {};
    if (wave == 0) {
        // Seed the bias into exactly one partial so the reduction is exact.
        const float br = b_hh[j];
        const float bz = b_hh[HIDDEN + j];
        const float bn = b_hh[2 * HIDDEN + j];
        #pragma unroll
        for (int i = 0; i < 8; ++i) { accr[i] = br; accz[i] = bz; accn[i] = bn; }
    }

    const __bf16* w_r = w_hh;
    const __bf16* w_z = w_hh + (size_t)HIDDEN * HIDDEN;
    const __bf16* w_n = w_hh + (size_t)2 * HIDDEN * HIDDEN;

    const int k0 = wave * KSPLIT;
    for (int k = k0; k < k0 + KSPLIT; k += 32) {
        v16bf af  = load_a_frag(h_bf, HIDDEN, row_base, k, lane);
        v16bf bfr = load_b_frag(w_r, HIDDEN, col_base, k, lane);
        v16bf bfz = load_b_frag(w_z, HIDDEN, col_base, k, lane);
        v16bf bfn = load_b_frag(w_n, HIDDEN, col_base, k, lane);
        accr = wmma_bf16(af, bfr, accr);
        accz = wmma_bf16(af, bfz, accz);
        accn = wmma_bf16(af, bfn, accn);
    }

    // Cross-wave reduction through LDS: waves 1..3 spill, wave 0 sums.
    __shared__ __attribute__((aligned(32))) float red[3 * 3 * 32 * 8];  // 9216 B
    if (wave != 0) {
        const int w = wave - 1;
        float* pr = &red[(((w * 3 + 0) * 32) + lane) * 8];
        float* pz = &red[(((w * 3 + 1) * 32) + lane) * 8];
        float* pn = &red[(((w * 3 + 2) * 32) + lane) * 8];
        *(v8f*)pr = accr;
        *(v8f*)pz = accz;
        *(v8f*)pn = accn;
    }
    __syncthreads();

    if (wave == 0) {
        #pragma unroll
        for (int w = 0; w < 3; ++w) {
            const v8f pr = *(const v8f*)&red[(((w * 3 + 0) * 32) + lane) * 8];
            const v8f pz = *(const v8f*)&red[(((w * 3 + 1) * 32) + lane) * 8];
            const v8f pn = *(const v8f*)&red[(((w * 3 + 2) * 32) + lane) * 8];
            #pragma unroll
            for (int i = 0; i < 8; ++i) { accr[i] += pr[i]; accz[i] += pz[i]; accn[i] += pn[i]; }
        }

        const int mrow = row_base + ((lane & 16) ? 8 : 0);
        #pragma unroll
        for (int v = 0; v < 8; ++v) {
            const int m = mrow + v;
            const size_t xrow = (size_t)m * (3 * HIDDEN);
            const float xr = (float)xp_t[xrow + j];
            const float xz = (float)xp_t[xrow + HIDDEN + j];
            const float xn = (float)xp_t[xrow + 2 * HIDDEN + j];
            const float r  = sigmoidf_fast(xr + accr[v]);
            const float z  = sigmoidf_fast(xz + accz[v]);
            const float nv = tanhf(xn + r * accn[v]);
            const float hp = h_f32[(size_t)m * HIDDEN + j];
            const float hn = (1.0f - z) * nv + z * hp;
            h_f32_out[(size_t)m * HIDDEN + j] = hn;
            const __bf16 hb = (__bf16)hn;
            h_bf_out[(size_t)m * HIDDEN + j] = hb;
            y_bf[(size_t)m * HIDDEN + j]     = hb;
        }
    }
}

// ---------------------------------------------------------------------------
// Small helper kernels
// ---------------------------------------------------------------------------
__global__ void cvt_f32_bf16(const float* __restrict__ src, __bf16* __restrict__ dst, size_t n) {
    size_t i = (size_t)blockIdx.x * blockDim.x + threadIdx.x;
    const size_t stride = (size_t)gridDim.x * blockDim.x;
    for (; i < n; i += stride) dst[i] = (__bf16)src[i];
}

__global__ void init_h_kernel(const float* __restrict__ src, float* __restrict__ hf,
                              __bf16* __restrict__ hb, size_t n) {
    size_t i = (size_t)blockIdx.x * blockDim.x + threadIdx.x;
    const size_t stride = (size_t)gridDim.x * blockDim.x;
    for (; i < n; i += stride) { const float v = src[i]; hf[i] = v; hb[i] = (__bf16)v; }
}

__global__ void copy_f32(const float* __restrict__ src, float* __restrict__ dst, size_t n) {
    size_t i = (size_t)blockIdx.x * blockDim.x + threadIdx.x;
    const size_t stride = (size_t)gridDim.x * blockDim.x;
    for (; i < n; i += stride) dst[i] = src[i];
}

// ---------------------------------------------------------------------------
// Host orchestration
// ---------------------------------------------------------------------------
extern "C" void kernel_launch(void* const* d_in, const int* in_sizes, int n_in,
                              void* d_out, int out_size, void* d_ws, size_t ws_size,
                              hipStream_t stream)
{
    (void)in_sizes; (void)n_in; (void)out_size; (void)ws_size;

    const float* inputs  = (const float*)d_in[0];
    const float* hidden0 = (const float*)d_in[1];
    const float* in_fc_w = (const float*)d_in[2];
    const float* in_fc_b = (const float*)d_in[3];
    const float* w_ih    = (const float*)d_in[4];
    const float* w_hh    = (const float*)d_in[5];
    const float* b_ih    = (const float*)d_in[6];
    const float* b_hh    = (const float*)d_in[7];
    const float* sm_fc_w = (const float*)d_in[8];
    const float* sm_fc_b = (const float*)d_in[9];
    float* out = (float*)d_out;

    const size_t M = (size_t)SEQ_LEN * BATCH;   // 65536 rows for batched GEMMs

    // Workspace carve-up (256B aligned slices)
    char* ws = (char*)d_ws;
    size_t off = 0;
    auto carve = [&](size_t bytes) -> char* {
        char* p = ws + off;
        off += (bytes + 255) & ~(size_t)255;
        return p;
    };
    __bf16* in_fc_w_bf = (__bf16*)carve((size_t)HIDDEN * INPUT_SZ * 2);
    __bf16* w_ih_bf    = (__bf16*)carve((size_t)LAYERS * 3 * HIDDEN * HIDDEN * 2);
    __bf16* w_hh_bf    = (__bf16*)carve((size_t)LAYERS * 3 * HIDDEN * HIDDEN * 2);
    __bf16* sm_fc_w_bf = (__bf16*)carve((size_t)INPUT_SZ * HIDDEN * 2);
    __bf16* in_bf      = (__bf16*)carve(M * INPUT_SZ * 2);
    __bf16* x_bf       = (__bf16*)carve(M * HIDDEN * 2);          // layer activations
    __bf16* xp_bf      = (__bf16*)carve(M * 3 * HIDDEN * 2);      // per-layer x-projection
    float*  h_f32[2];
    __bf16* h_bf[2];
    h_f32[0] = (float*) carve((size_t)BATCH * HIDDEN * 4);
    h_f32[1] = (float*) carve((size_t)BATCH * HIDDEN * 4);
    h_bf[0]  = (__bf16*)carve((size_t)BATCH * HIDDEN * 2);
    h_bf[1]  = (__bf16*)carve((size_t)BATCH * HIDDEN * 2);

    auto cvt = [&](const float* s, __bf16* d, size_t n) {
        int blocks = (int)((n + 255) / 256);
        if (blocks > 4096) blocks = 4096;
        cvt_f32_bf16<<<blocks, 256, 0, stream>>>(s, d, n);
    };
    auto gemm = [&](const __bf16* A, const __bf16* W, const float* bias,
                    __bf16* obf, float* of, int Mm, int Nn, int Kk) {
        const int tiles = (Mm >> 5) * (Nn >> 5);   // 32x32 per wave
        gemm_bias_wmma<<<tiles / 4, 128, 0, stream>>>(A, W, bias, obf, of, Mm, Nn, Kk);
    };

    // 1) One-time fp32 -> bf16 conversion of weights + inputs
    cvt(in_fc_w, in_fc_w_bf, (size_t)HIDDEN * INPUT_SZ);
    cvt(w_ih,    w_ih_bf,    (size_t)LAYERS * 3 * HIDDEN * HIDDEN);
    cvt(w_hh,    w_hh_bf,    (size_t)LAYERS * 3 * HIDDEN * HIDDEN);
    cvt(sm_fc_w, sm_fc_w_bf, (size_t)INPUT_SZ * HIDDEN);
    cvt(inputs,  in_bf,      M * INPUT_SZ);

    // 2) Input projection: x = inputs @ in_fc_w^T + b  -> bf16 [M,H]
    gemm(in_bf, in_fc_w_bf, in_fc_b, x_bf, nullptr, (int)M, HIDDEN, INPUT_SZ);

    // 3) Two GRU layers
    for (int l = 0; l < LAYERS; ++l) {
        // Batched input-side projection for the whole sequence: xp = x @ w_ih^T + b_ih
        gemm(x_bf, w_ih_bf + (size_t)l * 3 * HIDDEN * HIDDEN,
             b_ih + (size_t)l * 3 * HIDDEN, xp_bf, nullptr,
             (int)M, 3 * HIDDEN, HIDDEN);

        // h0 from the provided hidden state
        init_h_kernel<<<512, 256, 0, stream>>>(hidden0 + (size_t)l * BATCH * HIDDEN,
                                               h_f32[0], h_bf[0],
                                               (size_t)BATCH * HIDDEN);

        // Sequential recurrence (ping-pong h buffers)
        for (int t = 0; t < SEQ_LEN; ++t) {
            const int cur = t & 1, nxt = cur ^ 1;
            gru_step_wmma<<<512, 128, 0, stream>>>(
                h_bf[cur], h_f32[cur],
                xp_bf + (size_t)t * BATCH * 3 * HIDDEN,
                w_hh_bf + (size_t)l * 3 * HIDDEN * HIDDEN,
                b_hh + (size_t)l * 3 * HIDDEN,
                h_f32[nxt], h_bf[nxt],
                x_bf + (size_t)t * BATCH * HIDDEN);   // ys overwrites layer input
        }

        // SEQ_LEN even -> final state lands back in buffer 0
        copy_f32<<<512, 256, 0, stream>>>(h_f32[0],
                                          out + (size_t)SEQ_LEN * BATCH * INPUT_SZ
                                              + (size_t)l * BATCH * HIDDEN,
                                          (size_t)BATCH * HIDDEN);
    }

    // 4) Output projection: logits = x @ sm_fc_w^T + b -> fp32 into d_out
    gemm(x_bf, sm_fc_w_bf, sm_fc_b, nullptr, out, (int)M, INPUT_SZ, HIDDEN);
}
